// TFNN_83751862272707
// MI455X (gfx1250) — matrix operand order repaired and checked by reading, not compiled, not run on hardware
//
#include <hip/hip_runtime.h>
#include <hip/hip_bf16.h>
#include <math.h>

typedef __bf16 bf16;
typedef bf16  v16bf __attribute__((ext_vector_type(16)));
typedef bf16  v8bf  __attribute__((ext_vector_type(8)));
typedef bf16  v4bf  __attribute__((ext_vector_type(4)));
typedef float v8f   __attribute__((ext_vector_type(8)));

#define DH 128          // hidden width
#define TPB 256

// float atomic max via int/uint ordering trick (valid for mixed signs,
// target init value must be -inf = 0xff800000)
__device__ __forceinline__ void atomicMaxF(float* addr, float val) {
    if (val >= 0.0f) atomicMax((int*)addr, __float_as_int(val));
    else             atomicMin((unsigned int*)addr, __float_as_uint(val));
}

// ---------------- utility kernels ----------------
__global__ void fill_f32_kernel(float* __restrict__ p, float v, size_t n) {
    size_t i = (size_t)blockIdx.x * blockDim.x + threadIdx.x;
    if (i < n) p[i] = v;
}

__global__ void f32_to_bf16_kernel(const float* __restrict__ in,
                                   bf16* __restrict__ out, size_t n) {
    size_t i = (size_t)blockIdx.x * blockDim.x + threadIdx.x;
    if (i < n) out[i] = (bf16)in[i];
}

// h = relu(agg) in f32 and bf16 simultaneously (bf16 feeds next layer's GEMMs)
__global__ void relu_dual_kernel(const float* __restrict__ in,
                                 float* __restrict__ outf,
                                 bf16* __restrict__ outb, size_t n) {
    size_t i = (size_t)blockIdx.x * blockDim.x + threadIdx.x;
    if (i < n) {
        float v = in[i];
        v = v > 0.0f ? v : 0.0f;
        outf[i] = v;
        outb[i] = (bf16)v;
    }
}

// W[K x 128] (f32, row-major) -> WT[128 x Kpad] (bf16, row-major, zero-padded K)
__global__ void transpose_w_kernel(const float* __restrict__ W,
                                   bf16* __restrict__ WT, int K, int Kpad) {
    int idx = blockIdx.x * blockDim.x + threadIdx.x;
    if (idx >= DH * Kpad) return;
    int n  = idx / Kpad;
    int kk = idx - n * Kpad;
    WT[idx] = (kk < K) ? (bf16)W[kk * DH + n] : (bf16)0.0f;
}

// edge_attr[E x 16] f32 -> [E x 32] bf16 zero-padded (so K-loop is one 32-step)
__global__ void pad_edge_attr_kernel(const float* __restrict__ ea,
                                     bf16* __restrict__ out, size_t E) {
    size_t idx = (size_t)blockIdx.x * blockDim.x + threadIdx.x;
    if (idx >= E * 32) return;
    size_t e = idx >> 5;
    int kk = (int)(idx & 31);
    out[idx] = (kk < 16) ? (bf16)ea[e * 16 + kk] : (bf16)0.0f;
}

// ---------------- bf16 WMMA GEMM: C[M x 128] = A[M x K] * WT^T (+bias)(+relu) ----------------
// A row-major bf16 (lda = K, K in {32,64,128}); WT is [128 x K] bf16 row-major.
// One wave -> 16 rows x 128 cols. 8 waves/block -> 128 rows/block.
__global__ __launch_bounds__(TPB)
void gemm_bf16_n128_kernel(const bf16* __restrict__ A, const bf16* __restrict__ WT,
                           const float* __restrict__ bias,
                           float* __restrict__ Cf, bf16* __restrict__ Cbf,
                           int M, int K, int relu) {
    const int lane    = threadIdx.x & 31;
    const int wave    = threadIdx.x >> 5;
    const int rowBase = blockIdx.x * 128 + wave * 16;
    if (rowBase >= M) return;

    const int mrow = rowBase + (lane & 15);     // A row supplied by this lane
    const int klo  = (lane >> 4) << 3;          // 0 or 8 (A-frag half select)
    const int koff = (lane >> 4) << 3;          // 0 or 8 (B-frag half select)
    const int ncol = lane & 15;

    v8f zero = {0.f, 0.f, 0.f, 0.f, 0.f, 0.f, 0.f, 0.f};
    v8f acc[8];
#pragma unroll
    for (int t = 0; t < 8; ++t) acc[t] = zero;

    for (int kb = 0; kb < K; kb += 32) {
        // A fragment: lanes 0-15 hold K={kb..kb+7, kb+16..kb+23},
        //             lanes 16-31 hold K={kb+8..kb+15, kb+24..kb+31}
        const bf16* ap = A + (size_t)mrow * K + kb + klo;
        v8bf alo = *(const v8bf*)(ap);
        v8bf ahi = *(const v8bf*)(ap + 16);
        v16bf afrag = __builtin_shufflevector(alo, ahi,
            0, 1, 2, 3, 4, 5, 6, 7, 8, 9, 10, 11, 12, 13, 14, 15);
#pragma unroll
        for (int t = 0; t < 8; ++t) {
            // B fragment: lane n holds col (t*16+n), K in same order as A fragment
            const bf16* bp = WT + (size_t)(t * 16 + ncol) * K + kb + koff;
            v8bf blo = *(const v8bf*)(bp);
            v8bf bhi = *(const v8bf*)(bp + 16);
            v16bf bfrag = __builtin_shufflevector(blo, bhi,
                0, 1, 2, 3, 4, 5, 6, 7, 8, 9, 10, 11, 12, 13, 14, 15);
            acc[t] = __builtin_amdgcn_wmma_f32_16x16x32_bf16(
                false, afrag, false, bfrag, (short)0, acc[t], false, false);
        }
    }

    // C/D layout: vgpr r -> row rowBase + r + 8*(lane>=16), col = t*16 + (lane&15)
    const int crow = rowBase + ((lane >> 4) << 3);
#pragma unroll
    for (int t = 0; t < 8; ++t) {
        int col = t * 16 + ncol;
        float b = bias ? bias[col] : 0.0f;
#pragma unroll
        for (int r = 0; r < 8; ++r) {
            float v = acc[t][r] + b;
            if (relu) v = v > 0.0f ? v : 0.0f;
            size_t off = (size_t)(crow + r) * DH + col;
            if (Cf)  Cf[off]  = v;
            if (Cbf) Cbf[off] = (bf16)v;
        }
    }
}

// ---------------- edge pipeline ----------------
// pass 1: alpha[e] = q[dst].(k[src]+e)/sqrt(D); atomic segment-max into amax[dst]
__global__ __launch_bounds__(TPB)
void edge_alpha_kernel(const int* __restrict__ src, const int* __restrict__ dst,
                       const float* __restrict__ q, const float* __restrict__ k,
                       const bf16* __restrict__ ebuf,
                       float* __restrict__ alpha, float* __restrict__ amax, size_t E) {
    size_t edge = (size_t)blockIdx.x * 8 + (threadIdx.x >> 5);
    if (edge >= E) return;
    int lane = threadIdx.x & 31;
    int s = src[edge], d = dst[edge];
    float4 qv = *(const float4*)(q + (size_t)d * DH + lane * 4);
    float4 kv = *(const float4*)(k + (size_t)s * DH + lane * 4);
    v4bf   ev = *(const v4bf*)(ebuf + (size_t)edge * DH + lane * 4);
    float p = qv.x * (kv.x + (float)ev[0]) + qv.y * (kv.y + (float)ev[1]) +
              qv.z * (kv.z + (float)ev[2]) + qv.w * (kv.w + (float)ev[3]);
#pragma unroll
    for (int off = 16; off > 0; off >>= 1) p += __shfl_xor(p, off, 32);
    if (lane == 0) {
        float a = p * 0.08838834764831845f;   // 1/sqrt(128)
        alpha[edge] = a;
        atomicMaxF(&amax[d], a);
    }
}

// pass 2: alpha[e] <- exp(alpha[e] - amax[dst]); denom[dst] += exp(...)
__global__ void edge_exp_kernel(const int* __restrict__ dst,
                                float* __restrict__ alpha,
                                const float* __restrict__ amax,
                                float* __restrict__ denom, size_t E) {
    size_t e = (size_t)blockIdx.x * blockDim.x + threadIdx.x;
    if (e >= E) return;
    int d = dst[e];
    float ex = __expf(alpha[e] - amax[d]);
    alpha[e] = ex;
    atomicAdd(&denom[d], ex);
}

// pass 3: agg[dst] += (v[src]+e) * (ex/denom[dst])   (agg pre-seeded with h@ws+bs)
__global__ __launch_bounds__(TPB)
void edge_aggregate_kernel(const int* __restrict__ src, const int* __restrict__ dst,
                           const float* __restrict__ vbuf, const bf16* __restrict__ ebuf,
                           const float* __restrict__ alpha, const float* __restrict__ denom,
                           float* __restrict__ agg, size_t E) {
    size_t edge = (size_t)blockIdx.x * 8 + (threadIdx.x >> 5);
    if (edge >= E) return;
    int lane = threadIdx.x & 31;
    int s = src[edge], d = dst[edge];
    float w = alpha[edge] / denom[d];
    float4 vv = *(const float4*)(vbuf + (size_t)s * DH + lane * 4);
    v4bf   ev = *(const v4bf*)(ebuf + (size_t)edge * DH + lane * 4);
    float* ap = agg + (size_t)d * DH + lane * 4;
    atomicAdd(ap + 0, (vv.x + (float)ev[0]) * w);
    atomicAdd(ap + 1, (vv.y + (float)ev[1]) * w);
    atomicAdd(ap + 2, (vv.z + (float)ev[2]) * w);
    atomicAdd(ap + 3, (vv.w + (float)ev[3]) * w);
}

// ---------------- pooling + head ----------------
__global__ void pool_kernel(const float* __restrict__ h, const int* __restrict__ batch,
                            float* __restrict__ hg, float* __restrict__ cnt, size_t N) {
    size_t idx = (size_t)blockIdx.x * blockDim.x + threadIdx.x;
    if (idx >= N * 32) return;
    size_t node = idx >> 5;
    int j = (int)(idx & 31);
    int b = batch[node];
    float4 hv = *(const float4*)(h + node * DH + j * 4);
    float* gp = hg + (size_t)b * DH + j * 4;
    atomicAdd(gp + 0, hv.x);
    atomicAdd(gp + 1, hv.y);
    atomicAdd(gp + 2, hv.z);
    atomicAdd(gp + 3, hv.w);
    if (j == 0) atomicAdd(&cnt[b], 1.0f);
}

__global__ void head_kernel(const float* __restrict__ hg, const float* __restrict__ cnt,
                            const float* __restrict__ fc2_w, const float* __restrict__ fc2_b,
                            const float* __restrict__ fc3_w, const float* __restrict__ fc3_b,
                            float* __restrict__ out, int G) {
    int g = blockIdx.x * blockDim.x + threadIdx.x;
    if (g >= G) return;
    float c = cnt[g];
    c = c > 1.0f ? c : 1.0f;
    float inv = 1.0f / c;
    float r = fc3_b[0];
    for (int o = 0; o < 16; ++o) {
        float acc = fc2_b[o];
        for (int d2 = 0; d2 < DH; ++d2)
            acc += hg[(size_t)g * DH + d2] * inv * fc2_w[d2 * 16 + o];
        acc = acc > 0.0f ? acc : 0.0f;
        r += acc * fc3_w[o];
    }
    out[g] = r;
}

// ---------------- host-side orchestration ----------------
static inline unsigned nblk(size_t n, int t) { return (unsigned)((n + t - 1) / t); }

extern "C" void kernel_launch(void* const* d_in, const int* in_sizes, int n_in,
                              void* d_out, int out_size, void* d_ws, size_t ws_size,
                              hipStream_t stream) {
    const size_t N = (size_t)in_sizes[0] / 64;     // nodes
    const size_t E = (size_t)in_sizes[2] / 16;     // edges
    const int    G = out_size;                     // graphs (DIM_OUT=1)

    const float* x_in   = (const float*)d_in[0];
    const int*   ei     = (const int*)d_in[1];
    const int*   src    = ei;
    const int*   dst    = ei + E;
    const float* ea     = (const float*)d_in[2];
    const int*   batch  = (const int*)d_in[3];
    const float* fc1_w  = (const float*)d_in[4];
    const float* fc1_b  = (const float*)d_in[5];
    const float* fc2_w  = (const float*)d_in[33];
    const float* fc2_b  = (const float*)d_in[34];
    const float* fc3_w  = (const float*)d_in[35];
    const float* fc3_b  = (const float*)d_in[36];
    float* out = (float*)d_out;

    // ---- workspace carve-up (256B aligned) ----
    char* wp = (char*)d_ws;
    auto alloc = [&](size_t bytes) -> void* {
        void* p = (void*)wp;
        wp += (bytes + 255) & ~(size_t)255;
        return p;
    };
    float* h     = (float*)alloc(N * DH * 4);
    float* agg   = (float*)alloc(N * DH * 4);
    float* q     = (float*)alloc(N * DH * 4);
    float* k     = (float*)alloc(N * DH * 4);
    float* v     = (float*)alloc(N * DH * 4);
    bf16*  h_bf  = (bf16*)alloc(N * DH * 2);
    bf16*  x_bf  = (bf16*)alloc(N * 64 * 2);
    bf16*  ea_bf = (bf16*)alloc(E * 32 * 2);
    bf16*  e_bf  = (bf16*)alloc(E * DH * 2);
    float* alpha = (float*)alloc(E * 4);
    float* amax  = (float*)alloc(N * 4);
    float* denom = (float*)alloc(N * 4);
    float* hg    = (float*)alloc((size_t)G * DH * 4);
    float* cnt   = (float*)alloc((size_t)G * 4);
    bf16*  fc1T  = (bf16*)alloc((size_t)DH * 64 * 2);
    bf16*  wT[3][4];   // wq, wk, wv, ws  (each 128x128)
    bf16*  weT[3];     // 128x32 (zero-padded K=16)
    for (int l = 0; l < 3; ++l) {
        for (int t = 0; t < 4; ++t) wT[l][t] = (bf16*)alloc((size_t)DH * DH * 2);
        weT[l] = (bf16*)alloc((size_t)DH * 32 * 2);
    }
    (void)ws_size; (void)n_in;

    // ---- weight prep: transpose + convert to bf16 ----
    transpose_w_kernel<<<nblk(DH * 64, TPB), TPB, 0, stream>>>(fc1_w, fc1T, 64, 64);
    for (int l = 0; l < 3; ++l) {
        const int base = 6 + 9 * l;
        const float* wq = (const float*)d_in[base + 0];
        const float* wk = (const float*)d_in[base + 2];
        const float* wv = (const float*)d_in[base + 4];
        const float* we = (const float*)d_in[base + 6];
        const float* ws = (const float*)d_in[base + 7];
        transpose_w_kernel<<<nblk(DH * DH, TPB), TPB, 0, stream>>>(wq, wT[l][0], DH, DH);
        transpose_w_kernel<<<nblk(DH * DH, TPB), TPB, 0, stream>>>(wk, wT[l][1], DH, DH);
        transpose_w_kernel<<<nblk(DH * DH, TPB), TPB, 0, stream>>>(wv, wT[l][2], DH, DH);
        transpose_w_kernel<<<nblk(DH * DH, TPB), TPB, 0, stream>>>(ws, wT[l][3], DH, DH);
        transpose_w_kernel<<<nblk(DH * 32, TPB), TPB, 0, stream>>>(we, weT[l], 16, 32);
    }
    f32_to_bf16_kernel<<<nblk(N * 64, TPB), TPB, 0, stream>>>(x_in, x_bf, N * 64);
    pad_edge_attr_kernel<<<nblk(E * 32, TPB), TPB, 0, stream>>>(ea, ea_bf, E);

    // ---- fc1: h = x @ fc1_w + fc1_b  (writes f32 and bf16 copies) ----
    gemm_bf16_n128_kernel<<<nblk(N, 128), TPB, 0, stream>>>(
        x_bf, fc1T, fc1_b, h, h_bf, (int)N, 64, 0);

    // ---- 3 TransformerConv layers ----
    for (int l = 0; l < 3; ++l) {
        const int base = 6 + 9 * l;
        const float* bq = (const float*)d_in[base + 1];
        const float* bk = (const float*)d_in[base + 3];
        const float* bv = (const float*)d_in[base + 5];
        const float* bs = (const float*)d_in[base + 8];

        // q/k/v projections and skip term (skip seeds agg)
        gemm_bf16_n128_kernel<<<nblk(N, 128), TPB, 0, stream>>>(
            h_bf, wT[l][0], bq, q, (bf16*)nullptr, (int)N, DH, 0);
        gemm_bf16_n128_kernel<<<nblk(N, 128), TPB, 0, stream>>>(
            h_bf, wT[l][1], bk, k, (bf16*)nullptr, (int)N, DH, 0);
        gemm_bf16_n128_kernel<<<nblk(N, 128), TPB, 0, stream>>>(
            h_bf, wT[l][2], bv, v, (bf16*)nullptr, (int)N, DH, 0);
        gemm_bf16_n128_kernel<<<nblk(N, 128), TPB, 0, stream>>>(
            h_bf, wT[l][3], bs, agg, (bf16*)nullptr, (int)N, DH, 0);
        // edge projection e = edge_attr @ we  (bf16 output, no bias)
        gemm_bf16_n128_kernel<<<nblk(E, 128), TPB, 0, stream>>>(
            ea_bf, weT[l], (const float*)nullptr, (float*)nullptr, e_bf, (int)E, 32, 0);

        // segment softmax over incoming edges
        fill_f32_kernel<<<nblk(N, TPB), TPB, 0, stream>>>(amax, -INFINITY, N);
        fill_f32_kernel<<<nblk(N, TPB), TPB, 0, stream>>>(denom, 0.0f, N);
        edge_alpha_kernel<<<nblk(E * 32, TPB), TPB, 0, stream>>>(
            src, dst, q, k, e_bf, alpha, amax, E);
        edge_exp_kernel<<<nblk(E, TPB), TPB, 0, stream>>>(dst, alpha, amax, denom, E);
        edge_aggregate_kernel<<<nblk(E * 32, TPB), TPB, 0, stream>>>(
            src, dst, v, e_bf, alpha, denom, agg, E);

        // h = relu(agg), refresh bf16 copy for next layer's GEMMs
        relu_dual_kernel<<<nblk(N * DH, TPB), TPB, 0, stream>>>(agg, h, h_bf, N * DH);
    }

    // ---- global mean pool + MLP head ----
    fill_f32_kernel<<<nblk((size_t)G * DH, TPB), TPB, 0, stream>>>(hg, 0.0f, (size_t)G * DH);
    fill_f32_kernel<<<nblk((size_t)G, TPB), TPB, 0, stream>>>(cnt, 0.0f, (size_t)G);
    pool_kernel<<<nblk(N * 32, TPB), TPB, 0, stream>>>(h, batch, hg, cnt, N);
    head_kernel<<<nblk((size_t)G, TPB), TPB, 0, stream>>>(
        hg, cnt, fc2_w, fc2_b, fc3_w, fc3_b, out, G);
}
